// HeteroGNN_64845416235624
// MI455X (gfx1250) — matrix operand order repaired and checked by reading, not compiled, not run on hardware
//
#include <hip/hip_runtime.h>

// ---------------- types for WMMA ----------------
typedef __attribute__((ext_vector_type(16))) __bf16 v16bf;
typedef __attribute__((ext_vector_type(8)))  float  v8f;

union Frag {
    v16bf v;
    uint4 q[2];
};

__device__ __forceinline__ unsigned short f32_to_bf16(float f) {
    unsigned int u = __float_as_uint(f);
    unsigned int r = u + 0x7FFFu + ((u >> 16) & 1u);   // round-to-nearest-even
    return (unsigned short)(r >> 16);
}

// ---------------- constants ----------------
#define NBUS   100000
#define NGEN   20000
#define NLOAD  50000
#define NSHUNT 5000
#define FD     128
#define LDA    136          // padded LDS row stride (halfs): 272B, 16B-aligned, bank-spread
#define EAC 500000
#define ETR 100000
#define EGB 20000
#define EBG 20000
#define ELB 50000
#define EBL 50000
#define ESB 5000
#define EBS 5000

// ----------------------------------------------------------------------------
// GEMM: C[M,128] = A_bf16[M,128] @ Bt_bf16[128(out),128(in)] (+bias)(+relu)
// Block = 256 threads = 8 waves; block covers 128 rows x 128 cols.
//   * A stripe (128x128 bf16) staged once into LDS, shared by all 8 waves (8x dedup)
//   * each wave's 4 B fragments (its 16 columns, K=128) kept register-resident
//     and reused across 8 row tiles (8x dedup)
//   * 4 x v_wmma_f32_16x16x32_bf16 per 16x16 tile
// ----------------------------------------------------------------------------
__global__ void __launch_bounds__(256)
k_gemm_bf16(const unsigned short* __restrict__ A,
            const unsigned short* __restrict__ Bt,
            const float* __restrict__ bias,
            float* __restrict__ C,
            int M, int relu)
{
    __shared__ unsigned short lA[128 * LDA];   // 34 KB

    const int tid  = threadIdx.x;
    const int lane = tid & 31;
    const int wave = tid >> 5;          // 0..7 -> 16-column tile
    const int l15  = lane & 15;
    const int hs   = lane >> 4;         // K-half select
    const int row0 = blockIdx.x * 128;
    const int col  = wave * 16 + l15;

    // ---- B fragments: load once, keep in registers (columns disjoint per wave) ----
    const unsigned short* bp = Bt + (size_t)col * FD + hs * 16;
    Frag b0, b1, b2, b3;
    b0.q[0] = *(const uint4*)(bp +   0);  b0.q[1] = *(const uint4*)(bp +   8);
    b1.q[0] = *(const uint4*)(bp +  32);  b1.q[1] = *(const uint4*)(bp +  40);
    b2.q[0] = *(const uint4*)(bp +  64);  b2.q[1] = *(const uint4*)(bp +  72);
    b3.q[0] = *(const uint4*)(bp +  96);  b3.q[1] = *(const uint4*)(bp + 104);

    // ---- cooperative stage of the 128-row A stripe into LDS (row-clamped) ----
#pragma unroll
    for (int i = 0; i < 8; ++i) {
        int chunk = tid + i * 256;                 // 2048 x 16B chunks
        int r = chunk >> 4;                        // row within stripe
        int c = (chunk & 15) * 8;                  // half-column
        int gr = row0 + r;
        gr = (gr < M) ? gr : (M - 1);              // branchless clamp
        uint4 v = *(const uint4*)(A + (size_t)gr * FD + c);
        *(uint4*)(&lA[r * LDA + c]) = v;
    }
    __syncthreads();

    const float bv   = bias ? bias[col] : 0.0f;
    const bool  full = (row0 + 128) <= M;          // block-uniform -> scalar branch

#pragma unroll
    for (int rt = 0; rt < 8; ++rt) {
        const unsigned short* ap = &lA[(rt * 16 + l15) * LDA + hs * 16];
        Frag a0, a1, a2, a3;
        a0.q[0] = *(const uint4*)(ap +   0);  a0.q[1] = *(const uint4*)(ap +   8);
        a1.q[0] = *(const uint4*)(ap +  32);  a1.q[1] = *(const uint4*)(ap +  40);
        a2.q[0] = *(const uint4*)(ap +  64);  a2.q[1] = *(const uint4*)(ap +  72);
        a3.q[0] = *(const uint4*)(ap +  96);  a3.q[1] = *(const uint4*)(ap + 104);

        v8f acc = {0.f, 0.f, 0.f, 0.f, 0.f, 0.f, 0.f, 0.f};
        acc = __builtin_amdgcn_wmma_f32_16x16x32_bf16(false, a0.v, false, b0.v, (short)0, acc, false, false);
        acc = __builtin_amdgcn_wmma_f32_16x16x32_bf16(false, a1.v, false, b1.v, (short)0, acc, false, false);
        acc = __builtin_amdgcn_wmma_f32_16x16x32_bf16(false, a2.v, false, b2.v, (short)0, acc, false, false);
        acc = __builtin_amdgcn_wmma_f32_16x16x32_bf16(false, a3.v, false, b3.v, (short)0, acc, false, false);

        const int rbase = row0 + rt * 16 + hs * 8;
        if (full) {
#pragma unroll
            for (int j = 0; j < 8; ++j) {
                float v = acc[j] + bv;
                if (relu) v = fmaxf(v, 0.0f);
                C[(size_t)(rbase + j) * FD + col] = v;
            }
        } else {
#pragma unroll
            for (int j = 0; j < 8; ++j) {
                int r = rbase + j;
                if (r < M) {
                    float v = acc[j] + bv;
                    if (relu) v = fmaxf(v, 0.0f);
                    C[(size_t)r * FD + col] = v;
                }
            }
        }
    }
}

// ---------------- small helper kernels ----------------
__global__ void k_zero_f32(float* p, int n) {
    int i = blockIdx.x * blockDim.x + threadIdx.x;
    if (i < n) p[i] = 0.0f;
}

__global__ void k_to_bf16(const float* __restrict__ s, unsigned short* __restrict__ d, int n) {
    int i = blockIdx.x * blockDim.x + threadIdx.x;
    if (i < n) d[i] = f32_to_bf16(s[i]);
}

__global__ void k_relu_to_bf16(const float* __restrict__ h, unsigned short* __restrict__ x, int n) {
    int i = blockIdx.x * blockDim.x + threadIdx.x;
    if (i < n) x[i] = f32_to_bf16(fmaxf(h[i], 0.0f));
}

// transpose (and optionally sum up to 5) 128x128 f32 matrices -> bf16 [out][in]
__global__ void k_t5_bf16(const float* w0, const float* w1, const float* w2,
                          const float* w3, const float* w4, unsigned short* Wt)
{
    int i = blockIdx.x * blockDim.x + threadIdx.x;       // 16384
    if (i >= FD * FD) return;
    int n = i >> 7, k = i & 127;
    int src = k * FD + n;
    float s = w0[src];
    if (w1) s += w1[src];
    if (w2) s += w2[src];
    if (w3) s += w3[src];
    if (w4) s += w4[src];
    Wt[i] = f32_to_bf16(s);
}

__global__ void k_b5(const float* b0, const float* b1, const float* b2,
                     const float* b3, const float* b4, float* out)
{
    int i = threadIdx.x;
    if (i >= FD) return;
    float s = b0[i];
    if (b1) s += b1[i];
    if (b2) s += b2[i];
    if (b3) s += b3[i];
    if (b4) s += b4[i];
    out[i] = s;
}

__global__ void k_count(const long long* __restrict__ ei, float* __restrict__ cnt, int E) {
    int e = blockIdx.x * blockDim.x + threadIdx.x;
    if (e < E) atomicAdd(&cnt[(int)ei[E + e]], 1.0f);
}

__global__ void k_inv(float* c, int n) {
    int i = blockIdx.x * blockDim.x + threadIdx.x;
    if (i < n) c[i] = 1.0f / fmaxf(c[i], 1.0f);
}

// h[dst] += y[src] * inv[dst]; 32 threads per edge, 4 floats each
__global__ void k_scatter(const float* __restrict__ y, const long long* __restrict__ ei,
                          const float* __restrict__ inv, float* __restrict__ h, int E)
{
    int t = blockIdx.x * blockDim.x + threadIdx.x;
    int e = t >> 5;
    if (e >= E) return;
    int g = (t & 31) * 4;
    int s = (int)ei[e];
    int d = (int)ei[E + e];
    float sc = inv[d];
    float4 v = *reinterpret_cast<const float4*>(y + (size_t)s * FD + g);
    float* hp = h + (size_t)d * FD + g;
    atomicAdd(hp + 0, v.x * sc);
    atomicAdd(hp + 1, v.y * sc);
    atomicAdd(hp + 2, v.z * sc);
    atomicAdd(hp + 3, v.w * sc);
}

// out[i,0:2] = h1[i,:] @ W2[128,2] + b2
__global__ void k_head2(const float* __restrict__ h1, const float* __restrict__ W2,
                        const float* __restrict__ b2, float* __restrict__ out, int M)
{
    int i = blockIdx.x * blockDim.x + threadIdx.x;
    if (i >= M) return;
    const float* r = h1 + (size_t)i * FD;
    float s0 = b2[0], s1 = b2[1];
#pragma unroll 4
    for (int k = 0; k < FD; ++k) {
        float v = r[k];
        s0 += v * W2[k * 2 + 0];
        s1 += v * W2[k * 2 + 1];
    }
    out[i * 2 + 0] = s0;
    out[i * 2 + 1] = s1;
}

// ---------------- launch ----------------
static inline int cdiv(int a, int b) { return (a + b - 1) / b; }

extern "C" void kernel_launch(void* const* d_in, const int* in_sizes, int n_in,
                              void* d_out, int out_size, void* d_ws, size_t ws_size,
                              hipStream_t stream)
{
    (void)in_sizes; (void)n_in; (void)out_size; (void)ws_size;

    const float* x_bus   = (const float*)d_in[0];
    const float* x_gen   = (const float*)d_in[1];
    const float* x_load  = (const float*)d_in[2];
    const float* x_shunt = (const float*)d_in[3];
    const float* Wl  = (const float*)d_in[4];   // [3,8,128,128]
    const float* bl  = (const float*)d_in[5];   // [3,8,128]
    const float* Wr  = (const float*)d_in[6];   // [3,8,128,128]
    const float* Wb1 = (const float*)d_in[7];
    const float* bb1 = (const float*)d_in[8];
    const float* Wb2 = (const float*)d_in[9];
    const float* bb2 = (const float*)d_in[10];
    const float* Wg1 = (const float*)d_in[11];
    const float* bg1 = (const float*)d_in[12];
    const float* Wg2 = (const float*)d_in[13];
    const float* bg2 = (const float*)d_in[14];
    const long long* ei_ac = (const long long*)d_in[15];
    const long long* ei_tr = (const long long*)d_in[16];
    const long long* ei_gb = (const long long*)d_in[17];
    const long long* ei_bg = (const long long*)d_in[18];
    const long long* ei_lb = (const long long*)d_in[19];
    const long long* ei_bl = (const long long*)d_in[20];
    const long long* ei_sb = (const long long*)d_in[21];
    const long long* ei_bs = (const long long*)d_in[22];

    float* out = (float*)d_out;    // [100000*2] bus then [20000*2] gen

    // ---- workspace carve ----
    char* p = (char*)d_ws;
    auto alloc = [&](size_t bytes) -> void* {
        void* r = (void*)p;
        p += (bytes + 255) & ~(size_t)255;
        return r;
    };
    unsigned short* xb_bus   = (unsigned short*)alloc((size_t)NBUS   * FD * 2);
    unsigned short* xb_gen   = (unsigned short*)alloc((size_t)NGEN   * FD * 2);
    unsigned short* xb_load  = (unsigned short*)alloc((size_t)NLOAD  * FD * 2);
    unsigned short* xb_shunt = (unsigned short*)alloc((size_t)NSHUNT * FD * 2);
    float* h_bus   = (float*)alloc((size_t)NBUS   * FD * 4);
    float* h_gen   = (float*)alloc((size_t)NGEN   * FD * 4);
    float* h_load  = (float*)alloc((size_t)NLOAD  * FD * 4);
    float* h_shunt = (float*)alloc((size_t)NSHUNT * FD * 4);
    float* ybuf    = (float*)alloc((size_t)NBUS   * FD * 4);   // transform scratch (max src = bus)

    const int TOTINV = NBUS * 5 + NGEN + NLOAD + NSHUNT;       // 575000
    float* inv_all = (float*)alloc((size_t)TOTINV * 4);
    float* inv_ac = inv_all;
    float* inv_tr = inv_ac + NBUS;
    float* inv_gb = inv_tr + NBUS;
    float* inv_bg = inv_gb + NBUS;
    float* inv_lb = inv_bg + NGEN;
    float* inv_bl = inv_lb + NBUS;
    float* inv_sb = inv_bl + NLOAD;
    float* inv_bs = inv_sb + NBUS;

    unsigned short* Wlt[8];
    for (int i = 0; i < 8; ++i) Wlt[i] = (unsigned short*)alloc(FD * FD * 2);
    unsigned short* Wrt_bus   = (unsigned short*)alloc(FD * FD * 2);
    unsigned short* Wrt_gen   = (unsigned short*)alloc(FD * FD * 2);
    unsigned short* Wrt_load  = (unsigned short*)alloc(FD * FD * 2);
    unsigned short* Wrt_shunt = (unsigned short*)alloc(FD * FD * 2);
    unsigned short* Wht       = (unsigned short*)alloc(FD * FD * 2);   // head hidden weight
    float* bias_bus   = (float*)alloc(FD * 4);
    float* bias_gen   = (float*)alloc(FD * 4);
    float* bias_load  = (float*)alloc(FD * 4);
    float* bias_shunt = (float*)alloc(FD * 4);

    const int TPB = 256;
    const int WCNT = FD * FD;   // 16384

    // ---- 0. activations -> bf16 ----
    k_to_bf16<<<cdiv(NBUS   * FD, TPB), TPB, 0, stream>>>(x_bus,   xb_bus,   NBUS   * FD);
    k_to_bf16<<<cdiv(NGEN   * FD, TPB), TPB, 0, stream>>>(x_gen,   xb_gen,   NGEN   * FD);
    k_to_bf16<<<cdiv(NLOAD  * FD, TPB), TPB, 0, stream>>>(x_load,  xb_load,  NLOAD  * FD);
    k_to_bf16<<<cdiv(NSHUNT * FD, TPB), TPB, 0, stream>>>(x_shunt, xb_shunt, NSHUNT * FD);

    // ---- 1. per-edge-type inverse mean counts (layer invariant) ----
    k_zero_f32<<<cdiv(TOTINV, TPB), TPB, 0, stream>>>(inv_all, TOTINV);
    k_count<<<cdiv(EAC, TPB), TPB, 0, stream>>>(ei_ac, inv_ac, EAC);
    k_count<<<cdiv(ETR, TPB), TPB, 0, stream>>>(ei_tr, inv_tr, ETR);
    k_count<<<cdiv(EGB, TPB), TPB, 0, stream>>>(ei_gb, inv_gb, EGB);
    k_count<<<cdiv(EBG, TPB), TPB, 0, stream>>>(ei_bg, inv_bg, EBG);
    k_count<<<cdiv(ELB, TPB), TPB, 0, stream>>>(ei_lb, inv_lb, ELB);
    k_count<<<cdiv(EBL, TPB), TPB, 0, stream>>>(ei_bl, inv_bl, EBL);
    k_count<<<cdiv(ESB, TPB), TPB, 0, stream>>>(ei_sb, inv_sb, ESB);
    k_count<<<cdiv(EBS, TPB), TPB, 0, stream>>>(ei_bs, inv_bs, EBS);
    k_inv<<<cdiv(TOTINV, TPB), TPB, 0, stream>>>(inv_all, TOTINV);

    // ---- 2. layers ----
    for (int l = 0; l < 3; ++l) {
        const float* WlL = Wl + (size_t)l * 8 * FD * FD;
        const float* WrL = Wr + (size_t)l * 8 * FD * FD;
        const float* blL = bl + (size_t)l * 8 * FD;

        // per-edge-type message weights (transposed bf16)
        for (int et = 0; et < 8; ++et)
            k_t5_bf16<<<cdiv(WCNT, TPB), TPB, 0, stream>>>(
                WlL + (size_t)et * FD * FD, nullptr, nullptr, nullptr, nullptr, Wlt[et]);

        // combined root weights + biases per destination type
        k_t5_bf16<<<cdiv(WCNT, TPB), TPB, 0, stream>>>(
            WrL + 0 * FD * FD, WrL + 1 * FD * FD, WrL + 2 * FD * FD,
            WrL + 4 * FD * FD, WrL + 6 * FD * FD, Wrt_bus);
        k_t5_bf16<<<cdiv(WCNT, TPB), TPB, 0, stream>>>(
            WrL + 3 * FD * FD, nullptr, nullptr, nullptr, nullptr, Wrt_gen);
        k_t5_bf16<<<cdiv(WCNT, TPB), TPB, 0, stream>>>(
            WrL + 5 * FD * FD, nullptr, nullptr, nullptr, nullptr, Wrt_load);
        k_t5_bf16<<<cdiv(WCNT, TPB), TPB, 0, stream>>>(
            WrL + 7 * FD * FD, nullptr, nullptr, nullptr, nullptr, Wrt_shunt);
        k_b5<<<1, FD, 0, stream>>>(blL + 0 * FD, blL + 1 * FD, blL + 2 * FD,
                                   blL + 4 * FD, blL + 6 * FD, bias_bus);
        k_b5<<<1, FD, 0, stream>>>(blL + 3 * FD, nullptr, nullptr, nullptr, nullptr, bias_gen);
        k_b5<<<1, FD, 0, stream>>>(blL + 5 * FD, nullptr, nullptr, nullptr, nullptr, bias_load);
        k_b5<<<1, FD, 0, stream>>>(blL + 7 * FD, nullptr, nullptr, nullptr, nullptr, bias_shunt);

        // root terms: h = x_dst @ (sum Wr) + (sum bl)
        k_gemm_bf16<<<cdiv(NBUS,   128), TPB, 0, stream>>>(xb_bus,   Wrt_bus,   bias_bus,   h_bus,   NBUS,   0);
        k_gemm_bf16<<<cdiv(NGEN,   128), TPB, 0, stream>>>(xb_gen,   Wrt_gen,   bias_gen,   h_gen,   NGEN,   0);
        k_gemm_bf16<<<cdiv(NLOAD,  128), TPB, 0, stream>>>(xb_load,  Wrt_load,  bias_load,  h_load,  NLOAD,  0);
        k_gemm_bf16<<<cdiv(NSHUNT, 128), TPB, 0, stream>>>(xb_shunt, Wrt_shunt, bias_shunt, h_shunt, NSHUNT, 0);

        // message terms: transform-first (GEMM), then scaled scatter-add
        struct ET { const unsigned short* xs; int ns; const long long* ei; int E; float* inv; float* hd; };
        ET ets[8] = {
            { xb_bus,   NBUS,   ei_ac, EAC, inv_ac, h_bus   },  // et0: ac  bus->bus
            { xb_bus,   NBUS,   ei_tr, ETR, inv_tr, h_bus   },  // et1: tr  bus->bus
            { xb_gen,   NGEN,   ei_gb, EGB, inv_gb, h_bus   },  // et2: gb  gen->bus
            { xb_bus,   NBUS,   ei_bg, EBG, inv_bg, h_gen   },  // et3: bg  bus->gen
            { xb_load,  NLOAD,  ei_lb, ELB, inv_lb, h_bus   },  // et4: lb  load->bus
            { xb_bus,   NBUS,   ei_bl, EBL, inv_bl, h_load  },  // et5: bl  bus->load
            { xb_shunt, NSHUNT, ei_sb, ESB, inv_sb, h_bus   },  // et6: sb  shunt->bus
            { xb_bus,   NBUS,   ei_bs, EBS, inv_bs, h_shunt },  // et7: bs  bus->shunt
        };
        for (int et = 0; et < 8; ++et) {
            k_gemm_bf16<<<cdiv(ets[et].ns, 128), TPB, 0, stream>>>(
                ets[et].xs, Wlt[et], nullptr, ybuf, ets[et].ns, 0);
            k_scatter<<<cdiv(ets[et].E * 32, TPB), TPB, 0, stream>>>(
                ybuf, ets[et].ei, ets[et].inv, ets[et].hd, ets[et].E);
        }

        // ReLU -> next-layer bf16 activations
        k_relu_to_bf16<<<cdiv(NBUS   * FD, TPB), TPB, 0, stream>>>(h_bus,   xb_bus,   NBUS   * FD);
        k_relu_to_bf16<<<cdiv(NGEN   * FD, TPB), TPB, 0, stream>>>(h_gen,   xb_gen,   NGEN   * FD);
        k_relu_to_bf16<<<cdiv(NLOAD  * FD, TPB), TPB, 0, stream>>>(h_load,  xb_load,  NLOAD  * FD);
        k_relu_to_bf16<<<cdiv(NSHUNT * FD, TPB), TPB, 0, stream>>>(h_shunt, xb_shunt, NSHUNT * FD);
    }

    // ---- 3. heads ----
    // bus: relu(x @ Wb1 + bb1) @ Wb2 + bb2
    k_t5_bf16<<<cdiv(WCNT, TPB), TPB, 0, stream>>>(Wb1, nullptr, nullptr, nullptr, nullptr, Wht);
    k_gemm_bf16<<<cdiv(NBUS, 128), TPB, 0, stream>>>(xb_bus, Wht, bb1, h_bus, NBUS, 1);
    k_head2<<<cdiv(NBUS, TPB), TPB, 0, stream>>>(h_bus, Wb2, bb2, out, NBUS);
    // gen
    k_t5_bf16<<<cdiv(WCNT, TPB), TPB, 0, stream>>>(Wg1, nullptr, nullptr, nullptr, nullptr, Wht);
    k_gemm_bf16<<<cdiv(NGEN, 128), TPB, 0, stream>>>(xb_gen, Wht, bg1, h_gen, NGEN, 1);
    k_head2<<<cdiv(NGEN, TPB), TPB, 0, stream>>>(h_gen, Wg2, bg2, out + (size_t)NBUS * 2, NGEN);
}
